// AxonLIFNode_35983236006442
// MI455X (gfx1250) — compile-verified, matrix operand order checked by reading, not run on hardware
//
#include <hip/hip_runtime.h>

// AxonLIFNode on MI455X (gfx1250, wave32).
//   spikes: nonlinear LIF scan, computed per-lane (memory-bound, coalesced).
//   i_pot : linear axon filter == L(64x64, lower-tri powers of sigmoid(w)) @ spikes,
//           computed with V_WMMA_F32_16X16X32_F16 from LDS-staged f16 spikes.

typedef __attribute__((ext_vector_type(16))) _Float16 v16h;
typedef __attribute__((ext_vector_type(8)))  float    v8f;

#define BDIM 64
#define TDIM 64
#define NDIM 4096
#define V_TH_C 1.0f
#define INV_TAU_MEM 0.5f   // 1/TAU, TAU = 2
#define LPAD 66            // stride in halves: 33 dwords (odd) -> conflict-free LDS

__global__ __launch_bounds__(256)
void axon_lif_wmma_kernel(const float* __restrict__ X,
                          const float* __restrict__ wptr,
                          float* __restrict__ out_spk,
                          float* __restrict__ out_ipot) {
    // L: 64x64 lower-triangular filter matrix, f16, padded rows.
    __shared__ _Float16 Lmat[TDIM * LPAD];                 // 8448 B
    // S: per-wave spike tile, stored transposed [col][t], f16, padded.
    __shared__ _Float16 Ssp[8][32 * LPAD];                 // 8 * 4224 B

    const int tid  = threadIdx.x;
    const int wave = tid >> 5;
    const int lane = tid & 31;

    const float w       = wptr[0];
    const float inv_tau = 1.0f / (1.0f + __expf(-w));      // sigmoid(w); axon decay

    // Build L cooperatively: L[t][s] = inv_tau^(t-s) for s<=t, else 0.
    #pragma unroll
    for (int i = 0; i < 16; ++i) {
        const int idx = tid * 16 + i;                      // 4096 entries / 256 thr
        const int t   = idx >> 6;
        const int s   = idx & 63;
        const float v = (s <= t) ? __powf(inv_tau, (float)(t - s)) : 0.0f;
        Lmat[t * LPAD + s] = (_Float16)v;
    }

    // Block -> (batch row, 256-column slab of n). Wave -> 32 columns, lane -> 1 column.
    const int b     = blockIdx.x >> 4;
    const int nbase = (blockIdx.x & 15) << 8;
    const int col   = nbase + wave * 32 + lane;

    const float* Xp  = X       + ((size_t)b * TDIM) * NDIM + col;
    float*       spk = out_spk + ((size_t)b * TDIM) * NDIM + col;
    _Float16*   Scol = &Ssp[wave][lane * LPAD];

    // ---- Phase 1: sequential LIF scan over T (coalesced 128B/line per wave) ----
    float mem = 0.0f;
    #pragma unroll 8
    for (int t = 0; t < TDIM; ++t) {
        __builtin_prefetch(&Xp[(size_t)(t + 8) * NDIM], 0, 0);  // global_prefetch_b8
        const float x = Xp[(size_t)t * NDIM];
        mem += (x - mem) * INV_TAU_MEM;                    // mem + (x + Vreset - mem)/tau
        const float spike = (mem > V_TH_C) ? 1.0f : 0.0f;  // Heaviside(mem - v_th)
        mem = (1.0f - spike) * mem;                        // hard reset to 0
        spk[(size_t)t * NDIM] = spike;                     // coalesced f32 store
        Scol[t] = (_Float16)spike;                         // exact in f16 (0 or 1)
    }

    __syncthreads();  // Lmat ready for every wave (S tile is wave-private)

    // ---- Phase 2: i_pot tile = L @ S via WMMA ----
    const int m = lane & 15;   // row (A) / col (B) within 16-wide tile
    const int g = lane >> 4;   // lane group: K offset +8 per CDNA5 16-bit layout

    float* ipot = out_ipot + ((size_t)b * TDIM) * NDIM + nbase + wave * 32;

    #pragma unroll
    for (int tile = 0; tile < 2; ++tile) {                 // two 16-col tiles per wave
        const int tb = tile << 4;
        #pragma unroll
        for (int mblk = 0; mblk < 4; ++mblk) {             // output t-rows 16*mblk..+15
            v8f acc = {};
            #pragma unroll
            for (int kblk = 0; kblk < 2; ++kblk) {         // K = 64 -> 2 steps of 32
                if (kblk == 1 && mblk < 2) continue;       // all-zero A tile (upper tri)
                v16h a, bm;
                // CDNA5 16-bit fragment layout: VGPR p<4 -> K=2p,2p+1 ; p>=4 -> K=16+2(p-4),+1 ;
                // lanes 16-31 add +8 to K.
                #pragma unroll
                for (int h = 0; h < 16; ++h) {
                    const int p  = h >> 1;
                    const int lo = h & 1;
                    const int K  = ((p < 4) ? (p << 1) : (16 + ((p - 4) << 1)))
                                   + lo + (g << 3) + (kblk << 5);
                    a[h]  = Lmat[(mblk * 16 + m) * LPAD + K];       // A: L rows (M x K)
                    bm[h] = Ssp[wave][(tb + m) * LPAD + K];         // B: S cols (K x N)
                }
                acc = __builtin_amdgcn_wmma_f32_16x16x32_f16(
                          false, a, false, bm, (short)0, acc, false, false);
            }
            // D layout: VGPR r -> M=r (lanes 0-15) / M=r+8 (lanes 16-31), N = lane%16.
            #pragma unroll
            for (int r = 0; r < 8; ++r) {
                const int t = mblk * 16 + r + (g << 3);
                ipot[(size_t)t * NDIM + tb + m] = acc[r];
            }
        }
    }
}

extern "C" void kernel_launch(void* const* d_in, const int* in_sizes, int n_in,
                              void* d_out, int out_size, void* d_ws, size_t ws_size,
                              hipStream_t stream) {
    (void)in_sizes; (void)n_in; (void)out_size; (void)d_ws; (void)ws_size;
    const float* X = (const float*)d_in[0];
    const float* w = (const float*)d_in[1];
    float* out      = (float*)d_out;
    float* out_spk  = out;                                        // [B,T,N]
    float* out_ipot = out + (size_t)BDIM * TDIM * NDIM;           // [B,T,N]

    dim3 grid(BDIM * (NDIM / 256));  // 1024 blocks
    dim3 block(256);                 // 8 waves of 32
    axon_lif_wmma_kernel<<<grid, block, 0, stream>>>(X, w, out_spk, out_ipot);
}
